// MambaBlock_62277025792206
// MI455X (gfx1250) — matrix operand order repaired and checked by reading, not compiled, not run on hardware
//
#include <hip/hip_runtime.h>
#include <hip/hip_bf16.h>

#define D_MODEL 1024
#define D_STATE 16
#define D_CONV  4
#define DIM     2048
#define BSZ     8
#define SEQ_L   2048
#define SEQ_LC  2051   // L + D_CONV - 1

typedef __attribute__((ext_vector_type(16))) __bf16 v16bf;
typedef __attribute__((ext_vector_type(8)))  float  v8f;

// ---------------------------------------------------------------------------
// WMMA GEMM:  C(MxN) = A(MxK) * W(NxK)^T + bias(N)
// fp32 in/out, bf16 WMMA f32-accumulate (v_wmma_f32_16x16x32_bf16).
// Block = 256 threads (8 wave32 waves), block tile 128(M) x 128(N), K-step 64
// (two 32-wide K subtiles -> 16 WMMAs per iteration per wave).
// Software pipeline: register-staged global loads + double-buffered LDS so
// global load latency overlaps the WMMA work; one barrier per K-step.
// LDS slabs are pre-arranged in WMMA fragment order so each lane's fragment
// is one contiguous 32-byte LDS read (2 x ds_load_b128).
// ---------------------------------------------------------------------------
__global__ __launch_bounds__(256)
void gemm_bf16_wmma(const float* __restrict__ A, const float* __restrict__ W,
                    const float* __restrict__ bias, float* __restrict__ C,
                    int M, int N, int K)
{
    // [buf][kt*8 + tile][lane][16 halves]
    __shared__ alignas(32) __bf16 As[2][16 * 512];
    __shared__ alignas(32) __bf16 Bs[2][16 * 512];

    const int tid  = threadIdx.x;
    const int lane = tid & 31;
    const int wave = tid >> 5;
    const int wm   = wave >> 2;   // 0..1 : wave m-quad (4 m-tiles)
    const int wn   = wave & 3;    // 0..3 : wave n-pair (2 n-tiles)

    const int m0 = blockIdx.x * 128;
    const int n0 = blockIdx.y * 128;

    const int lr0 = tid >> 4;     // base row (0..15), rows lr0 + 16*it
    const int c4  = tid & 15;     // float4 column within 64-wide K slab

    float4 ra[8], rb[8];

    auto load_regs = [&](int k0) {
        #pragma unroll
        for (int it = 0; it < 8; ++it) {
            int r = lr0 + it * 16;
            int grow = m0 + r;
            ra[it] = make_float4(0.f, 0.f, 0.f, 0.f);
            if (grow < M)
                ra[it] = *(const float4*)(A + (size_t)grow * K + k0 + 4 * c4);
            int gn = n0 + r;
            rb[it] = make_float4(0.f, 0.f, 0.f, 0.f);
            if (gn < N)
                rb[it] = *(const float4*)(W + (size_t)gn * K + k0 + 4 * c4);
        }
    };

    auto store_lds = [&](int buf) {
        #pragma unroll
        for (int it = 0; it < 8; ++it) {
            int r  = lr0 + it * 16;
            int mm = r & 15;
            int tt = r >> 4;
            const float* pa = &ra[it].x;
            const float* pb = &rb[it].x;
            #pragma unroll
            for (int j = 0; j < 4; ++j) {
                int k  = 4 * c4 + j;                       // 0..63
                int kt = k >> 5;                           // K subtile
                int kl = k & 31;
                // A fragment order (ISA 16-bit A 16x32 layout)
                int ha = (kl >> 3) & 1;
                int ea = (kl & 7) | ((kl & 16) >> 1);
                As[buf][(kt * 8 + tt) * 512 + (16 * ha + mm) * 16 + ea] = (__bf16)pa[j];
                // B fragment order (same as A: ISA 16-bit 32x16 layout)
                int hb = (kl >> 3) & 1;
                int eb = (kl & 7) | ((kl & 16) >> 1);
                Bs[buf][(kt * 8 + tt) * 512 + (16 * hb + mm) * 16 + eb] = (__bf16)pb[j];
            }
        }
    };

    v8f acc[4][2] = {};

    // pipeline prologue
    load_regs(0);
    store_lds(0);
    __syncthreads();

    int cur = 0;
    for (int k0 = 0; k0 < K; k0 += 64) {
        const bool hasNext = (k0 + 64) < K;
        if (hasNext) load_regs(k0 + 64);     // issue global loads early

        #pragma unroll
        for (int kt = 0; kt < 2; ++kt) {
            const v16bf* Ap = (const v16bf*)&As[cur][kt * 8 * 512];
            const v16bf* Bp = (const v16bf*)&Bs[cur][kt * 8 * 512];
            v16bf af[4], bfr[2];
            #pragma unroll
            for (int i = 0; i < 4; ++i) af[i]  = Ap[(4 * wm + i) * 32 + lane];
            #pragma unroll
            for (int j = 0; j < 2; ++j) bfr[j] = Bp[(2 * wn + j) * 32 + lane];

            #pragma unroll
            for (int i = 0; i < 4; ++i)
                #pragma unroll
                for (int j = 0; j < 2; ++j)
                    acc[i][j] = __builtin_amdgcn_wmma_f32_16x16x32_bf16(
                        false, af[i], false, bfr[j], (short)0, acc[i][j], false, false);
        }

        if (hasNext) store_lds(cur ^ 1);     // consume loads after matrix work
        __syncthreads();
        cur ^= 1;
    }

    // ---- store D (ISA C/D layout: VGPR r -> M = r + 8*(lane>=16), N = lane&15) ----
    const int nlo = lane & 15;
    const int hh  = lane >> 4;
    #pragma unroll
    for (int i = 0; i < 4; ++i) {
        #pragma unroll
        for (int j = 0; j < 2; ++j) {
            int nb = n0 + (2 * wn + j) * 16 + nlo;
            float bv = (bias != nullptr && nb < N) ? bias[nb] : 0.f;
            #pragma unroll
            for (int r = 0; r < 8; ++r) {
                int mb = m0 + (4 * wm + i) * 16 + r + 8 * hh;
                if (mb < M && nb < N)
                    C[(size_t)mb * N + nb] = acc[i][j][r] + bv;
            }
        }
    }
}

// ---------------------------------------------------------------------------
// Depthwise padded conv1d: xc[b,t,d] = cb[d] + sum_k cw[d,k]*xp[b,t-3+k,d]
// t in [0, SEQ_LC); xp rows valid in [0, SEQ_L). Memory-bound, coalesced in d.
// ---------------------------------------------------------------------------
__global__ __launch_bounds__(256)
void conv_dw(const float* __restrict__ xp, const float* __restrict__ cw,
             const float* __restrict__ cb, float* __restrict__ xc)
{
    size_t idx = (size_t)blockIdx.x * 256 + threadIdx.x;
    const size_t total = (size_t)BSZ * SEQ_LC * DIM;
    if (idx >= total) return;
    int d = (int)(idx % DIM);
    int t = (int)((idx / DIM) % SEQ_LC);
    int b = (int)(idx / ((size_t)DIM * SEQ_LC));
    float acc = cb[d];
    #pragma unroll
    for (int k = 0; k < D_CONV; ++k) {
        int ti = t - (D_CONV - 1) + k;
        if (ti >= 0 && ti < SEQ_L)
            acc += cw[d * D_CONV + k] * xp[((size_t)b * SEQ_L + ti) * DIM + d];
    }
    xc[idx] = acc;
}

// ---------------------------------------------------------------------------
// Fused delta + selective scan. One thread per (b, d) channel; 16 states in
// registers; x_dbl rows (48 floats/t) staged through LDS in chunks of 32 t.
// delta = softplus(b_dt[d] + d_raw . W_dt[d,:]) computed on the fly (K=16).
// ---------------------------------------------------------------------------
__global__ __launch_bounds__(256)
void scan_kernel(const float* __restrict__ xc, const float* __restrict__ xdbl,
                 const float* __restrict__ W_dt, const float* __restrict__ b_dt,
                 float* __restrict__ y)
{
    const int CT = 32;
    __shared__ float sl[CT * 48];
    const int tid = threadIdx.x;
    const int d = blockIdx.x * 256 + tid;
    const int b = blockIdx.y;

    float w[D_STATE];
    #pragma unroll
    for (int n = 0; n < D_STATE; ++n) w[n] = W_dt[(size_t)d * D_STATE + n];
    const float bd = b_dt[d];
    float Ak[D_STATE];
    #pragma unroll
    for (int n = 0; n < D_STATE; ++n)
        Ak[n] = -__expf(-(5.0f / 15.0f) * (float)n);   // A = -exp(linspace(0,-5,16))
    float h[D_STATE];
    #pragma unroll
    for (int n = 0; n < D_STATE; ++n) h[n] = 0.f;

    const size_t rowbase = (size_t)b * SEQ_LC;
    for (int tc = 0; tc < SEQ_LC; tc += CT) {
        int cnt = (SEQ_LC - tc < CT) ? (SEQ_LC - tc) : CT;
        __syncthreads();
        for (int i = tid; i < cnt * 48; i += 256)
            sl[i] = xdbl[(rowbase + tc) * 48 + i];
        __syncthreads();
        for (int tt = 0; tt < cnt; ++tt) {
            const float* row = &sl[tt * 48];      // [0:16)=d_raw [16:32)=B [32:48)=C
            const size_t ro = (rowbase + tc + tt) * DIM + d;
            float xr = xc[ro];
            float dr = bd;
            #pragma unroll
            for (int n = 0; n < D_STATE; ++n) dr += row[n] * w[n];
            float delta = (dr > 20.f) ? dr : __logf(1.f + __expf(dr));
            float yv = 0.f;
            #pragma unroll
            for (int n = 0; n < D_STATE; ++n) {
                h[n] = h[n] * __expf(delta * Ak[n]) + xr * row[16 + n];
                yv  += h[n] * row[32 + n];
            }
            y[ro] = yv;
        }
    }
}

// ---------------------------------------------------------------------------
extern "C" void kernel_launch(void* const* d_in, const int* in_sizes, int n_in,
                              void* d_out, int out_size, void* d_ws, size_t ws_size,
                              hipStream_t stream) {
    const float* x     = (const float*)d_in[0];
    const float* W_in  = (const float*)d_in[1];
    const float* b_in  = (const float*)d_in[2];
    const float* cw    = (const float*)d_in[3];
    const float* cb    = (const float*)d_in[4];
    const float* W_x   = (const float*)d_in[5];
    const float* W_dt  = (const float*)d_in[6];
    const float* b_dt  = (const float*)d_in[7];
    const float* W_out = (const float*)d_in[8];
    const float* b_out = (const float*)d_in[9];
    float* out = (float*)d_out;

    float* ws = (float*)d_ws;
    const size_t rows = (size_t)BSZ * SEQ_LC;           // 16408
    float* xc   = ws;                                   // rows*DIM
    float* buf1 = xc + rows * DIM;                      // xp (16384 rows) then y (rows)
    float* xdbl = buf1 + rows * DIM;                    // rows*48

    // 1) xp = x @ W_in^T + b_in     (M=16384, N=2048, K=1024)
    {
        dim3 g((BSZ * SEQ_L + 127) / 128, (DIM + 127) / 128);
        gemm_bf16_wmma<<<g, 256, 0, stream>>>(x, W_in, b_in, buf1,
                                              BSZ * SEQ_L, DIM, D_MODEL);
    }
    // 2) depthwise conv -> xc        (B, Lc, DIM)
    {
        size_t total = rows * DIM;
        conv_dw<<<(unsigned)((total + 255) / 256), 256, 0, stream>>>(buf1, cw, cb, xc);
    }
    // 3) x_dbl = xc @ W_x^T          (M=16408, N=48, K=2048)
    {
        dim3 g(((int)rows + 127) / 128, (3 * D_STATE + 127) / 128);
        gemm_bf16_wmma<<<g, 256, 0, stream>>>(xc, W_x, nullptr, xdbl,
                                              (int)rows, 3 * D_STATE, DIM);
    }
    // 4) fused delta + selective scan -> y in buf1 (xp is dead now)
    {
        dim3 g(DIM / 256, BSZ);
        scan_kernel<<<g, 256, 0, stream>>>(xc, xdbl, W_dt, b_dt, buf1);
    }
    // 5) out = y @ W_out^T + b_out   (M=16408, N=1024, K=2048)
    {
        dim3 g(((int)rows + 127) / 128, (D_MODEL + 127) / 128);
        gemm_bf16_wmma<<<g, 256, 0, stream>>>(buf1, W_out, b_out, out,
                                              (int)rows, D_MODEL, DIM);
    }
}
